// TwoLayerGAT_75977971466896
// MI455X (gfx1250) — matrix operand order repaired and checked by reading, not compiled, not run on hardware
//
#include <hip/hip_runtime.h>

// ---------------------------------------------------------------------------
// Two-layer GAT for MI455X (gfx1250, wave32).
// GEMMs use v_wmma_f32_16x16x32_bf16 (f32 accumulate); edge softmax +
// aggregation use encoded-float atomicMax and global f32 atomicAdd.
// ---------------------------------------------------------------------------

typedef __attribute__((ext_vector_type(16))) __bf16 v16bf;
typedef __attribute__((ext_vector_type(8)))  float  v8f;

#define NEG_SLOPE 0.2f

// ---- monotone float <-> uint encoding for atomic segment-max --------------
__device__ __forceinline__ unsigned enc_f32(float f) {
    unsigned u = __float_as_uint(f);
    return (u & 0x80000000u) ? ~u : (u | 0x80000000u);
}
__device__ __forceinline__ float dec_f32(unsigned u) {
    unsigned v = (u & 0x80000000u) ? (u & 0x7fffffffu) : ~u;
    return __uint_as_float(v);
}

// ---------------------------------------------------------------------------
// Pre-swizzle W [K, NC] fp32 row-major into per-lane WMMA B fragments (bf16).
// Fragment (kstep, nt): lane l (half = l>>4, n = nt*16 + (l&15)) holds
// 16 bf16 with k = kstep*32 + (j<8 ? half*8 + j : 16 + half*8 + j-8)  -> one 32B chunk.
// ---------------------------------------------------------------------------
__global__ void swizzle_w_kernel(const float* __restrict__ W,
                                 __bf16* __restrict__ Wfrag, int K, int NC) {
    int idx = blockIdx.x * blockDim.x + threadIdx.x;
    int ntiles = NC >> 4;
    int total = (K >> 5) * ntiles * 32;
    if (idx >= total) return;
    int lane  = idx & 31;
    int t     = idx >> 5;
    int nt    = t % ntiles;
    int kstep = t / ntiles;
    int half  = lane >> 4;
    int n     = nt * 16 + (lane & 15);
    __bf16* dst = Wfrag + (size_t)idx * 16;
#pragma unroll
    for (int j = 0; j < 16; ++j) {
        int k = kstep * 32 + (j < 8 ? half * 8 + j : 16 + half * 8 + (j - 8));
        dst[j] = (__bf16)W[(size_t)k * NC + n];
    }
}

// ---------------------------------------------------------------------------
// Y[16-row strip] = X[strip, K] @ W  via v_wmma_f32_16x16x32_bf16.
// Block = (NC/16) waves; wave nt owns output tile (blockIdx.x*16, nt*16).
// A fp32 -> bf16 converted in-register; B read pre-swizzled.
// ---------------------------------------------------------------------------
__global__ void gemm_bf16_wmma_kernel(const float* __restrict__ X,
                                      const __bf16* __restrict__ Wfrag,
                                      float* __restrict__ Y, int K, int NC) {
    const int lane   = threadIdx.x & 31;
    const int nt     = threadIdx.x >> 5;   // wave id == N-tile
    const int ntiles = NC >> 4;
    const int half   = lane >> 4;
    const int m      = blockIdx.x * 16 + (lane & 15);
    const float* arow = X + (size_t)m * K;

    v8f c = {};
    for (int kk = 0; kk < K; kk += 32) {
        const float* ap = arow + kk + half * 8;
        float4 a0 = *(const float4*)(ap);
        float4 a1 = *(const float4*)(ap + 4);
        float4 a2 = *(const float4*)(ap + 16);
        float4 a3 = *(const float4*)(ap + 20);
        v16bf av;
        av[0]  = (__bf16)a0.x; av[1]  = (__bf16)a0.y;
        av[2]  = (__bf16)a0.z; av[3]  = (__bf16)a0.w;
        av[4]  = (__bf16)a1.x; av[5]  = (__bf16)a1.y;
        av[6]  = (__bf16)a1.z; av[7]  = (__bf16)a1.w;
        av[8]  = (__bf16)a2.x; av[9]  = (__bf16)a2.y;
        av[10] = (__bf16)a2.z; av[11] = (__bf16)a2.w;
        av[12] = (__bf16)a3.x; av[13] = (__bf16)a3.y;
        av[14] = (__bf16)a3.z; av[15] = (__bf16)a3.w;

        const int kstep = kk >> 5;
        v16bf bv = *(const v16bf*)(Wfrag +
                     ((size_t)(kstep * ntiles + nt) * 32 + lane) * 16);

        c = __builtin_amdgcn_wmma_f32_16x16x32_bf16(
                /*neg_a=*/false, av, /*neg_b=*/false, bv,
                /*c_mod=*/(short)0, c, /*reuse_a=*/false, /*reuse_b=*/false);
    }

    const int n = nt * 16 + (lane & 15);
    float* yrow = Y + (size_t)(blockIdx.x * 16 + half * 8) * NC + n;
#pragma unroll
    for (int r = 0; r < 8; ++r) yrow[(size_t)r * NC] = c[r];
}

// ---------------------------------------------------------------------------
// el[n,h] = <feat[n,h,:], al[h,:]> ; er likewise. One thread per (n,h).
// ---------------------------------------------------------------------------
__global__ void node_attn_kernel(const float* __restrict__ feat,
                                 const float* __restrict__ al,
                                 const float* __restrict__ ar,
                                 float* __restrict__ el, float* __restrict__ er,
                                 int N, int H, int D) {
    int idx = blockIdx.x * blockDim.x + threadIdx.x;
    if (idx >= N * H) return;
    int h = idx % H;
    int n = idx / H;
    const float* f = feat + (size_t)n * H * D + h * D;
    const float* a = al + h * D;
    const float* b = ar + h * D;
    float sl = 0.f, sr = 0.f;
#pragma unroll 8
    for (int d = 0; d < D; ++d) { float v = f[d]; sl += v * a[d]; sr += v * b[d]; }
    el[idx] = sl;
    er[idx] = sr;
}

// ---------------------------------------------------------------------------
// Pass A: per-edge LeakyReLU logit + encoded atomic segment-max over dst.
// ---------------------------------------------------------------------------
__global__ void edge_logits_max_kernel(const int* __restrict__ src,
                                       const int* __restrict__ dst,
                                       const float* __restrict__ el,
                                       const float* __restrict__ er,
                                       float* __restrict__ elog,
                                       unsigned* __restrict__ mx, int E, int H) {
    int idx = blockIdx.x * blockDim.x + threadIdx.x;
    if (idx >= E * H) return;
    int h = idx % H;
    int e = idx / H;
    int s = src[e], d = dst[e];
    float lg = el[s * H + h] + er[d * H + h];
    lg = lg > 0.f ? lg : NEG_SLOPE * lg;
    elog[idx] = lg;
    atomicMax(&mx[d * H + h], enc_f32(lg));
}

// Decode encoded maxima in place; untouched (zero-in-degree) -> 0.0.
__global__ void decode_max_kernel(unsigned* __restrict__ mx, int n) {
    int idx = blockIdx.x * blockDim.x + threadIdx.x;
    if (idx >= n) return;
    unsigned u = mx[idx];
    float m = (u == 0u) ? 0.f : dec_f32(u);
    ((float*)mx)[idx] = m;
}

// ---------------------------------------------------------------------------
// Pass B: ee = exp(logit - m[dst]); overwrite elog; atomic segment-sum into z.
// ---------------------------------------------------------------------------
__global__ void edge_exp_sum_kernel(const int* __restrict__ dst,
                                    float* __restrict__ elog,
                                    const float* __restrict__ m,
                                    float* __restrict__ z, int E, int H) {
    int idx = blockIdx.x * blockDim.x + threadIdx.x;
    if (idx >= E * H) return;
    int h = idx % H;
    int e = idx / H;
    int d = dst[e];
    float ee = __expf(elog[idx] - m[d * H + h]);
    elog[idx] = ee;
    atomicAdd(&z[d * H + h], ee);
}

// Initialize accumulator rows with the bias vector (so rst + b is free).
__global__ void init_bias_kernel(float* __restrict__ out,
                                 const float* __restrict__ b, int N, int HD) {
    int idx = blockIdx.x * blockDim.x + threadIdx.x;
    if (idx >= N * HD) return;
    out[idx] = b[idx % HD];
}

// ---------------------------------------------------------------------------
// Pass C: rst[dst,h,d] += feat[src,h,d] * (ee / max(z[dst,h], 1e-16)).
// One thread per (edge, h, d) -> coalesced f32 atomic adds.
// ---------------------------------------------------------------------------
__global__ void edge_aggregate_kernel(const int* __restrict__ src,
                                      const int* __restrict__ dst,
                                      const float* __restrict__ feat,
                                      const float* __restrict__ ee,
                                      const float* __restrict__ z,
                                      float* __restrict__ rst,
                                      int E, int H, int D) {
    int idx = blockIdx.x * blockDim.x + threadIdx.x;
    int HD = H * D;
    if (idx >= E * HD) return;
    int r = idx % HD;
    int e = idx / HD;
    int h = r / D;
    int s  = src[e];
    int d0 = dst[e];
    float zz = z[d0 * H + h];
    zz = zz > 1e-16f ? zz : 1e-16f;
    float a = ee[e * H + h] / zz;
    atomicAdd(&rst[(size_t)d0 * HD + r], feat[(size_t)s * HD + r] * a);
}

// ---------------------------------------------------------------------------
// Host-side launcher
// ---------------------------------------------------------------------------
extern "C" void kernel_launch(void* const* d_in, const int* in_sizes, int n_in,
                              void* d_out, int out_size, void* d_ws, size_t ws_size,
                              hipStream_t stream) {
    // setup_inputs() dict order:
    const float* x   = (const float*)d_in[0];   // [N, 256]
    const float* W1  = (const float*)d_in[1];   // [256, 128]
    const float* al1 = (const float*)d_in[2];   // [4, 32]
    const float* ar1 = (const float*)d_in[3];
    const float* b1  = (const float*)d_in[4];   // [128]
    const float* W2  = (const float*)d_in[5];   // [128, 32]
    const float* al2 = (const float*)d_in[6];   // [1, 32]
    const float* ar2 = (const float*)d_in[7];
    const float* b2  = (const float*)d_in[8];   // [32]
    const int*   src = (const int*)d_in[9];     // [E]
    const int*   dst = (const int*)d_in[10];    // [E]
    float* out = (float*)d_out;                 // [N, 32]

    const int F_IN = 256;
    const int H1 = 4, D1 = 32, HD1 = H1 * D1;   // 128
    const int H2 = 1, D2 = 32, HD2 = H2 * D2;   // 32
    const int N = in_sizes[0] / F_IN;
    const int E = in_sizes[9];

    // ---- workspace carve-out (256B aligned slices) ----
    size_t off = 0;
    auto alloc = [&](size_t bytes) -> void* {
        off = (off + 255) & ~(size_t)255;
        void* p = (char*)d_ws + off;
        off += bytes;
        return p;
    };
    __bf16*   W1f   = (__bf16*)alloc((size_t)(F_IN / 32) * (HD1 / 16) * 512 * sizeof(__bf16));
    __bf16*   W2f   = (__bf16*)alloc((size_t)(HD1 / 32) * (HD2 / 16) * 512 * sizeof(__bf16));
    float*    feat1 = (float*)alloc((size_t)N * HD1 * sizeof(float));
    float*    hbuf  = (float*)alloc((size_t)N * HD1 * sizeof(float));
    float*    feat2 = (float*)alloc((size_t)N * HD2 * sizeof(float));
    float*    el1   = (float*)alloc((size_t)N * H1 * sizeof(float));
    float*    er1   = (float*)alloc((size_t)N * H1 * sizeof(float));
    unsigned* mx1   = (unsigned*)alloc((size_t)N * H1 * sizeof(unsigned));
    float*    z1    = (float*)alloc((size_t)N * H1 * sizeof(float));
    float*    elog1 = (float*)alloc((size_t)E * H1 * sizeof(float));
    float*    el2   = (float*)alloc((size_t)N * H2 * sizeof(float));
    float*    er2   = (float*)alloc((size_t)N * H2 * sizeof(float));
    unsigned* mx2   = (unsigned*)alloc((size_t)N * H2 * sizeof(unsigned));
    float*    z2    = (float*)alloc((size_t)N * H2 * sizeof(float));
    float*    elog2 = (float*)alloc((size_t)E * H2 * sizeof(float));
    (void)ws_size; (void)n_in; (void)out_size;

    const int TB = 256;
    auto blocks = [&](long long n) { return (int)((n + TB - 1) / TB); };

    // zero segment-max / segment-sum accumulators (graph-capture safe)
    hipMemsetAsync(mx1, 0, (size_t)N * H1 * sizeof(unsigned), stream);
    hipMemsetAsync(z1,  0, (size_t)N * H1 * sizeof(float),    stream);
    hipMemsetAsync(mx2, 0, (size_t)N * H2 * sizeof(unsigned), stream);
    hipMemsetAsync(z2,  0, (size_t)N * H2 * sizeof(float),    stream);

    // weight swizzle into WMMA B fragments
    swizzle_w_kernel<<<blocks((F_IN / 32) * (HD1 / 16) * 32), TB, 0, stream>>>(W1, W1f, F_IN, HD1);
    swizzle_w_kernel<<<blocks((HD1 / 32) * (HD2 / 16) * 32), TB, 0, stream>>>(W2, W2f, HD1, HD2);

    // ================= Layer 1 =================
    gemm_bf16_wmma_kernel<<<N / 16, (HD1 / 16) * 32, 0, stream>>>(x, W1f, feat1, F_IN, HD1);
    node_attn_kernel<<<blocks((long long)N * H1), TB, 0, stream>>>(feat1, al1, ar1, el1, er1, N, H1, D1);
    edge_logits_max_kernel<<<blocks((long long)E * H1), TB, 0, stream>>>(src, dst, el1, er1, elog1, mx1, E, H1);
    decode_max_kernel<<<blocks((long long)N * H1), TB, 0, stream>>>(mx1, N * H1);
    edge_exp_sum_kernel<<<blocks((long long)E * H1), TB, 0, stream>>>(dst, elog1, (const float*)mx1, z1, E, H1);
    init_bias_kernel<<<blocks((long long)N * HD1), TB, 0, stream>>>(hbuf, b1, N, HD1);
    edge_aggregate_kernel<<<blocks((long long)E * HD1), TB, 0, stream>>>(src, dst, feat1, elog1, z1, hbuf, E, H1, D1);

    // ================= Layer 2 =================
    gemm_bf16_wmma_kernel<<<N / 16, (HD2 / 16) * 32, 0, stream>>>(hbuf, W2f, feat2, HD1, HD2);
    node_attn_kernel<<<blocks((long long)N * H2), TB, 0, stream>>>(feat2, al2, ar2, el2, er2, N, H2, D2);
    edge_logits_max_kernel<<<blocks((long long)E * H2), TB, 0, stream>>>(src, dst, el2, er2, elog2, mx2, E, H2);
    decode_max_kernel<<<blocks((long long)N * H2), TB, 0, stream>>>(mx2, N * H2);
    edge_exp_sum_kernel<<<blocks((long long)E * H2), TB, 0, stream>>>(dst, elog2, (const float*)mx2, z2, E, H2);
    init_bias_kernel<<<blocks((long long)N * HD2), TB, 0, stream>>>(out, b2, N, HD2);
    edge_aggregate_kernel<<<blocks((long long)E * HD2), TB, 0, stream>>>(src, dst, feat2, elog2, z2, out, E, H2, D2);
}
